// naiveTreeLSTM_4294967296281
// MI455X (gfx1250) — matrix operand order repaired and checked by reading, not compiled, not run on hardware
//
#include <hip/hip_runtime.h>
#include <hip/hip_bf16.h>
#include <stdint.h>

// ---------------------------------------------------------------------------
// TreeLSTM on MI455X (gfx1250): level-sequential fused WMMA kernels.
//   gates = [x | (h_l + h_r)] @ [W_ih | W_hh]^T   (K = 384, N = 1024)
// bf16 operands, f32 accumulation via v_wmma_f32_16x16x32_bf16.
// Each wave owns 16 output columns across ALL four gates -> pointwise LSTM
// math fused in registers, gates never written to memory.
// Per K-chunk both the A tile (64x32) and the block's weight slice (128x32)
// are staged in LDS, so the inner loop issues only the A global stream;
// B fragments come from ds_load_b128 (4-wave dedup of weight fragments).
// Workspace usage: ~97 MB (bf16 ping-pong h/c buffers + packed weights).
// ---------------------------------------------------------------------------

typedef __attribute__((ext_vector_type(16))) __bf16 v16bf;
typedef __attribute__((ext_vector_type(8)))  float  v8f;

#define IN_DIM      128
#define MEM_DIM     256
#define NUM_CLASSES 12
#define NUM_LEVELS  17
#define KDIM        (IN_DIM + MEM_DIM)   // 384
#define MT          64                   // node rows per block
#define KT          32                   // K chunk (one bf16 WMMA K)
#define WROWS       128                  // gate-rows per block (4 gates x 32)

__device__ __forceinline__ unsigned short f2bf(float f) {
    union { float f; unsigned u; } v; v.f = f;
    unsigned u = v.u;
    u += 0x7FFFu + ((u >> 16) & 1u);     // round-to-nearest-even
    return (unsigned short)(u >> 16);
}
__device__ __forceinline__ float bf2f(unsigned short b) {
    union { unsigned u; float f; } v; v.u = ((unsigned)b) << 16;
    return v.f;
}

struct Pack32 { uint4 lo, hi; };
__device__ __forceinline__ v16bf pack_frag(uint4 lo, uint4 hi) {
    Pack32 p; p.lo = lo; p.hi = hi;
    return __builtin_bit_cast(v16bf, p);
}

__device__ __forceinline__ float sigmoidf(float x) {
    return 1.0f / (1.0f + expf(-x));
}

// ---------------------------------------------------------------------------
// Pack W = [W_ih | W_hh] as bf16 [1024][384], and bc = b_ih + b_hh (f32).
// ---------------------------------------------------------------------------
__global__ void prep_weights(const float* __restrict__ W_ih,
                             const float* __restrict__ W_hh,
                             const float* __restrict__ b_ih,
                             const float* __restrict__ b_hh,
                             unsigned short* __restrict__ Wb,
                             float* __restrict__ bc)
{
    int idx = blockIdx.x * blockDim.x + threadIdx.x;
    if (idx < 4 * MEM_DIM * KDIM) {
        int r = idx / KDIM;
        int k = idx - r * KDIM;
        float v = (k < IN_DIM) ? W_ih[r * IN_DIM + k]
                               : W_hh[r * MEM_DIM + (k - IN_DIM)];
        Wb[idx] = f2bf(v);
    }
    if (idx < 4 * MEM_DIM) bc[idx] = b_ih[idx] + b_hh[idx];
}

// ---------------------------------------------------------------------------
// One tree level. Block: 256 threads = 8 waves; tile MT=64 nodes x 32 cols.
// Wave w: mtile = w>>1 (16-row subtile), colg = w&1 (16-col group).
// Each wave accumulates 4 gate tiles (i,f,g,o) of 16x16 in v8f accumulators.
// ---------------------------------------------------------------------------
__global__ __launch_bounds__(256)
void tree_lstm_level(const float* __restrict__ x,
                     const unsigned short* __restrict__ Wb,    // [1024][384] bf16
                     const float* __restrict__ bc,             // [1024]
                     const unsigned short* __restrict__ hprev, // [2n][256] bf16 or null
                     const unsigned short* __restrict__ cprev, // [2n][256] bf16 or null
                     unsigned short* __restrict__ hcur,        // [n][256] bf16
                     unsigned short* __restrict__ ccur,        // [n][256] bf16
                     int levelBase, int n)
{
    __shared__ __align__(16) unsigned short As[MT * KT];       // 4 KB A chunk
    __shared__ __align__(16) unsigned short Wsh[WROWS * KT];   // 8 KB W chunk

    const int tid   = threadIdx.x;
    const int wave  = tid >> 5;
    const int lane  = tid & 31;
    const int mtile = wave >> 1;          // 0..3
    const int colg  = wave & 1;           // 0..1
    const int rowBase = blockIdx.x * MT;
    const int colB    = blockIdx.y * 32;              // block's 32 h-cols
    const int colBase = colB + colg * 16;             // this wave's 16 h-cols
    const int khalf = lane >> 4;
    const int l16   = lane & 15;

    const v8f vzero = {0.f,0.f,0.f,0.f,0.f,0.f,0.f,0.f};
    v8f acc[4] = {vzero, vzero, vzero, vzero};        // i, f, g, o

    for (int kb = 0; kb < KDIM / KT; ++kb) {
        __syncthreads();
        // -------- cooperative A-chunk load: As[m][k], k in [kb*32, kb*32+32)
        {
            const int linear = tid * 8;               // 2048 elems / 256 thr
            const int m  = linear >> 5;
            const int k0 = linear & 31;
            const int gm = rowBase + m;
            const int kk = kb * KT + k0;
            unsigned short vals[8];
            if (gm < n) {
                if (kk < IN_DIM) {
                    const float4* xr =
                        (const float4*)(x + (size_t)(levelBase + gm) * IN_DIM + kk);
                    float4 a = xr[0], b = xr[1];
                    vals[0]=f2bf(a.x); vals[1]=f2bf(a.y); vals[2]=f2bf(a.z); vals[3]=f2bf(a.w);
                    vals[4]=f2bf(b.x); vals[5]=f2bf(b.y); vals[6]=f2bf(b.z); vals[7]=f2bf(b.w);
                } else if (hprev != nullptr) {
                    const int kh = kk - IN_DIM;
                    const unsigned short* hl = hprev + (size_t)(2*gm)     * MEM_DIM + kh;
                    const unsigned short* hr = hprev + (size_t)(2*gm + 1) * MEM_DIM + kh;
                    #pragma unroll
                    for (int e = 0; e < 8; ++e)
                        vals[e] = f2bf(bf2f(hl[e]) + bf2f(hr[e]));   // child h-sum fused
                } else {
                    #pragma unroll
                    for (int e = 0; e < 8; ++e) vals[e] = 0;
                }
            } else {
                #pragma unroll
                for (int e = 0; e < 8; ++e) vals[e] = 0;
            }
            #pragma unroll
            for (int e = 0; e < 8; ++e) As[linear + e] = vals[e];
        }
        // -------- cooperative W-chunk load: Wsh[rr][k], rr = g*32 + c
        {
            const int linear = tid * 16;              // 4096 elems / 256 thr
            const int rr = linear >> 5;               // 0..127
            const int k0 = linear & 31;               // 0 or 16
            const int g  = rr >> 5;
            const int c  = rr & 31;
            const int r  = g * MEM_DIM + colB + c;    // weight (gate) row
            const unsigned short* wsrc = Wb + (size_t)r * KDIM + kb * KT + k0;
            __builtin_prefetch(wsrc + KT, 0, 1);      // next K chunk
            const uint4 w0 = *(const uint4*)(wsrc);
            const uint4 w1 = *(const uint4*)(wsrc + 8);
            *(uint4*)(Wsh + linear)     = w0;
            *(uint4*)(Wsh + linear + 8) = w1;
        }
        __syncthreads();

        // -------- A fragment (16x32 bf16, ISA 7.12.2 layout)
        const unsigned short* arow = As + (mtile * 16 + l16) * KT;
        const uint4 alo = *(const uint4*)(arow + khalf * 8);
        const uint4 ahi = *(const uint4*)(arow + 16 + khalf * 8);
        const v16bf afrag = pack_frag(alo, ahi);

        // -------- B fragments (one per gate) from LDS weight slice
        #pragma unroll
        for (int g = 0; g < 4; ++g) {
            const unsigned short* wrow =
                Wsh + (g * 32 + colg * 16 + l16) * KT;
            const uint4 blo = *(const uint4*)(wrow + khalf * 8);
            const uint4 bhi = *(const uint4*)(wrow + 16 + khalf * 8);
            const v16bf bfrag = pack_frag(blo, bhi);
            acc[g] = __builtin_amdgcn_wmma_f32_16x16x32_bf16(
                         false, afrag, false, bfrag,
                         (short)0, acc[g], false, false);
        }
    }

    // -------- fused pointwise LSTM cell (all 4 gates resident per wave)
    const int col = colBase + l16;
    const float bi = bc[0 * MEM_DIM + col];
    const float bf = bc[1 * MEM_DIM + col];
    const float bg = bc[2 * MEM_DIM + col];
    const float bo = bc[3 * MEM_DIM + col];

    #pragma unroll
    for (int v = 0; v < 8; ++v) {
        const int m = rowBase + mtile * 16 + khalf * 8 + v;  // C/D: M = v + 8*(lane>=16)
        if (m < n) {
            const float gi = acc[0][v] + bi;
            const float gf = acc[1][v] + bf;
            const float gg = acc[2][v] + bg;
            const float go = acc[3][v] + bo;
            float chc = 0.f;
            if (hprev != nullptr) {
                chc = bf2f(cprev[(size_t)(2*m)     * MEM_DIM + col]) +
                      bf2f(cprev[(size_t)(2*m + 1) * MEM_DIM + col]);
            }
            const float cn = sigmoidf(gf) * chc + sigmoidf(gi) * tanhf(gg);
            const float hn = sigmoidf(go) * tanhf(cn);
            hcur[(size_t)m * MEM_DIM + col] = f2bf(hn);
            ccur[(size_t)m * MEM_DIM + col] = f2bf(cn);
        }
    }
}

// ---------------------------------------------------------------------------
// Final FC: out[j] = c_root @ W_fc[j] + b_fc[j], 12x256 — one wave.
// ---------------------------------------------------------------------------
__global__ void fc_head(const unsigned short* __restrict__ c_root,
                        const float* __restrict__ W_fc,
                        const float* __restrict__ b_fc,
                        float* __restrict__ out)
{
    const int j = threadIdx.x;
    if (j < NUM_CLASSES) {
        float s = b_fc[j];
        for (int k = 0; k < MEM_DIM; ++k)
            s += bf2f(c_root[k]) * W_fc[j * MEM_DIM + k];
        out[j] = s;
    }
}

// ---------------------------------------------------------------------------
extern "C" void kernel_launch(void* const* d_in, const int* in_sizes, int n_in,
                              void* d_out, int out_size, void* d_ws, size_t ws_size,
                              hipStream_t stream)
{
    (void)in_sizes; (void)n_in; (void)out_size; (void)ws_size;
    const float* x    = (const float*)d_in[0];
    const float* W_ih = (const float*)d_in[1];
    const float* b_ih = (const float*)d_in[2];
    const float* W_hh = (const float*)d_in[3];
    const float* b_hh = (const float*)d_in[4];
    const float* W_fc = (const float*)d_in[5];
    const float* b_fc = (const float*)d_in[6];
    // d_in[7] = num_levels (device scalar); fixed at compile time = 17.

    char* ws = (char*)d_ws;
    // Workspace layout (bytes, 256-aligned). Total ~97 MB.
    const size_t szWb = (size_t)4 * MEM_DIM * KDIM * 2;          //   768 KB
    const size_t szBc = (size_t)4 * MEM_DIM * 4;                 //     4 KB
    const size_t szH0 = (size_t)65536 * MEM_DIM * 2;             //    32 MB (levels 16,14,...)
    const size_t szH1 = (size_t)32768 * MEM_DIM * 2;             //    16 MB (levels 15,13,...)
    unsigned short* Wb = (unsigned short*)(ws);
    float*          bc = (float*)(ws + szWb);
    unsigned short* h0 = (unsigned short*)(ws + szWb + szBc);
    unsigned short* h1 = (unsigned short*)(ws + szWb + szBc + szH0);
    unsigned short* c0 = (unsigned short*)(ws + szWb + szBc + szH0 + szH1);
    unsigned short* c1 = (unsigned short*)(ws + szWb + szBc + szH0 + szH1 + szH0);

    prep_weights<<<(4 * MEM_DIM * KDIM + 255) / 256, 256, 0, stream>>>(
        W_ih, W_hh, b_ih, b_hh, Wb, bc);

    for (int l = NUM_LEVELS - 1; l >= 0; --l) {
        const int n = 1 << l;
        const int cur = (NUM_LEVELS - 1 - l) & 1;                // level-16 -> buf 0
        unsigned short* hcur = cur ? h1 : h0;
        unsigned short* ccur = cur ? c1 : c0;
        const unsigned short* hprev =
            (l == NUM_LEVELS - 1) ? nullptr : (cur ? h0 : h1);
        const unsigned short* cprev =
            (l == NUM_LEVELS - 1) ? nullptr : (cur ? c0 : c1);
        dim3 grid((unsigned)((n + MT - 1) / MT), MEM_DIM / 32);  // (nodes/64, 8)
        tree_lstm_level<<<grid, 256, 0, stream>>>(
            x, Wb, bc, hprev, cprev, hcur, ccur, n - 1, n);
    }

    // Level 0 parity: (16-0)&1 == 0 -> root c lives in c0 row 0.
    fc_head<<<1, 32, 0, stream>>>(c0, W_fc, b_fc, (float*)d_out);
}
